// TwoDTPERoPEAttention_24713241821815
// MI455X (gfx1250) — compile-verified, matrix operand order checked by reading, not run on hardware
//
#include <hip/hip_runtime.h>

#define BATCH 2
#define SEQ   2048
#define DMODEL 512
#define NHEAD 8
#define HDIM  64
#define MROWS (BATCH*SEQ)   // 4096

#if defined(__gfx1250__) && \
    __has_builtin(__builtin_amdgcn_global_load_async_to_lds_b128) && \
    __has_builtin(__builtin_amdgcn_s_wait_asynccnt)
#define USE_ASYNC 1
#else
#define USE_ASYNC 0
#endif

typedef __attribute__((ext_vector_type(16))) __bf16 v16bf;
typedef __attribute__((ext_vector_type(8)))  float  v8f;

__device__ __forceinline__ unsigned short f2bf(float f) {
  unsigned u = __builtin_bit_cast(unsigned, f);
  unsigned r = u + 0x7FFFu + ((u >> 16) & 1u);   // round-to-nearest-even
  return (unsigned short)(r >> 16);
}

typedef union { v16bf v; uint4 u[2]; } fragU;

// Load a WMMA 16-bit fragment: halves 0-7 from p0, halves 8-15 from p1
// (each 8 consecutive bf16 = one 16-byte load). Works for global and LDS.
__device__ __forceinline__ v16bf ld_frag(const unsigned short* p0,
                                         const unsigned short* p1) {
  fragU f;
  f.u[0] = *(const uint4*)p0;
  f.u[1] = *(const uint4*)p1;
  return f.v;
}

#if USE_ASYNC
typedef int v4i __attribute__((ext_vector_type(4)));
typedef __attribute__((address_space(1))) v4i* gp128;   // global int4*
typedef __attribute__((address_space(3))) v4i* lp128;   // LDS int4*

// Stage one 32x64 bf16 K-block (4 KB) into LDS with 8 async b128 copies:
// lane l copies row k0+l; the immediate offset applies to BOTH addresses.
__device__ __forceinline__ void stage_k_async(const unsigned short* Kp, int k0,
                                              unsigned short* dst, int lane) {
  gp128 g = (gp128)(Kp + (size_t)(k0 + lane) * HDIM);
  lp128 l = (lp128)(dst + lane * HDIM);
  __builtin_amdgcn_global_load_async_to_lds_b128(g, l,   0, 0);
  __builtin_amdgcn_global_load_async_to_lds_b128(g, l,  16, 0);
  __builtin_amdgcn_global_load_async_to_lds_b128(g, l,  32, 0);
  __builtin_amdgcn_global_load_async_to_lds_b128(g, l,  48, 0);
  __builtin_amdgcn_global_load_async_to_lds_b128(g, l,  64, 0);
  __builtin_amdgcn_global_load_async_to_lds_b128(g, l,  80, 0);
  __builtin_amdgcn_global_load_async_to_lds_b128(g, l,  96, 0);
  __builtin_amdgcn_global_load_async_to_lds_b128(g, l, 112, 0);
}
#endif

// ---------------------------------------------------------------------------
// One-time conversions: f32 -> bf16 and f32 -> bf16 transposed.
// ---------------------------------------------------------------------------
__global__ __launch_bounds__(256) void cvt_bf16(const float* __restrict__ in,
                                                unsigned short* __restrict__ out,
                                                int n) {
  int i = blockIdx.x * 256 + threadIdx.x;
  if (i < n) out[i] = f2bf(in[i]);
}

__global__ __launch_bounds__(256) void cvt_bf16_t(const float* __restrict__ W,
                                                  unsigned short* __restrict__ Wt,
                                                  int K, int N) {
  int i = blockIdx.x * 256 + threadIdx.x;   // i = n*K + k (coalesced writes)
  if (i >= K * N) return;
  int k = i % K, n = i / K;
  Wt[i] = f2bf(W[(size_t)k * N + n]);
}

// ---------------------------------------------------------------------------
// C[M,N] = A[M,K] @ Wt[N,K]^T + bias ; bf16 WMMA, f32 accumulate.
// One wave computes a 32x64 tile: 2 A-frags x 4 B-frags -> 8 WMMAs per K-step.
// ---------------------------------------------------------------------------
__global__ __launch_bounds__(256) void gemm_bias_wmma(
    const unsigned short* __restrict__ A, const unsigned short* __restrict__ Wt,
    const float* __restrict__ bias, float* __restrict__ C,
    int Mr, int N, int K)
{
  int wid  = (int)((blockIdx.x * blockDim.x + threadIdx.x) >> 5);
  int lane = threadIdx.x & 31;
  int ntiles = N >> 6;             // 64-wide N tiles
  int tm = (wid / ntiles) << 5;    // 32 rows
  int tn = (wid % ntiles) << 6;    // 64 cols
  if (tm >= Mr) return;
  int g8 = (lane >> 4) << 3;
  int r  = lane & 15;

  v8f z = {};
  v8f acc[2][4];
#pragma unroll
  for (int i = 0; i < 2; ++i)
#pragma unroll
    for (int c = 0; c < 4; ++c) acc[i][c] = z;

  const unsigned short* a0p = A + (size_t)(tm + r) * K;
  const unsigned short* a1p = A + (size_t)(tm + 16 + r) * K;

  for (int kb = 0; kb < K; kb += 32) {
    __builtin_prefetch(a0p + kb + 128, 0, 1);   // global_prefetch_b8
    v16bf a0 = ld_frag(a0p + kb + g8, a0p + kb + 16 + g8);
    v16bf a1 = ld_frag(a1p + kb + g8, a1p + kb + 16 + g8);
#pragma unroll
    for (int c = 0; c < 4; ++c) {
      const unsigned short* wp = Wt + (size_t)(tn + c * 16 + r) * K + kb;
      v16bf bf = ld_frag(wp + g8, wp + 16 + g8);
      acc[0][c] = __builtin_amdgcn_wmma_f32_16x16x32_bf16(false, a0, false, bf,
                                                          (short)0, acc[0][c], false, false);
      acc[1][c] = __builtin_amdgcn_wmma_f32_16x16x32_bf16(false, a1, false, bf,
                                                          (short)0, acc[1][c], false, false);
    }
  }
#pragma unroll
  for (int c = 0; c < 4; ++c) {
    float bv = bias[tn + c * 16 + r];
#pragma unroll
    for (int i = 0; i < 2; ++i)
#pragma unroll
      for (int rr = 0; rr < 8; ++rr)
        C[(size_t)(tm + i * 16 + g8 + rr) * N + tn + c * 16 + r] = acc[i][c][rr] + bv;
  }
}

// ---------------------------------------------------------------------------
// Stable counting sort of positions (values in [0,SEQ)).
// ---------------------------------------------------------------------------
__global__ __launch_bounds__(256) void sort_positions(
    const int* __restrict__ pr, const int* __restrict__ pc,
    int* __restrict__ sortbuf)
{
  __shared__ int hist[SEQ];
  int set = blockIdx.x >> 1;
  int b   = blockIdx.x & 1;
  const int* pos = (set ? pc : pr) + b * SEQ;
  int* out = sortbuf + (size_t)(set * 2 + b) * SEQ;
  for (int i = threadIdx.x; i < SEQ; i += blockDim.x) hist[i] = 0;
  __syncthreads();
  for (int i = threadIdx.x; i < SEQ; i += blockDim.x) atomicAdd(&hist[pos[i]], 1);
  __syncthreads();
  if (threadIdx.x == 0) {
    int run = 0;
    for (int i = 0; i < SEQ; ++i) { int c = hist[i]; hist[i] = run; run += c; }
    for (int l = 0; l < SEQ; ++l) { int p = pos[l]; out[hist[p]] = l; hist[p]++; }
  }
}

// ---------------------------------------------------------------------------
// RoPE + gather into sorted order. Emits:
//   qs/ks : bf16 [set][b][h][j][d]   (row-major, WMMA A/B friendly)
//   vt    : bf16 [set][b][h][d][j]   (transposed: PV B-frag contiguous)
// ---------------------------------------------------------------------------
__global__ __launch_bounds__(256) void rope_gather(
    const float* __restrict__ q, const float* __restrict__ k,
    const float* __restrict__ v, const int* __restrict__ pr,
    const int* __restrict__ pc, const int* __restrict__ sortbuf,
    unsigned short* __restrict__ qs, unsigned short* __restrict__ ks,
    unsigned short* __restrict__ vt)
{
  size_t t = (size_t)blockIdx.x * 256 + threadIdx.x;
  int d = (int)(t & 63);      size_t t1 = t >> 6;
  int j = (int)(t1 & 2047);   size_t t2 = t1 >> 11;
  int h = (int)(t2 & 7);      size_t t3 = t2 >> 3;
  int b = (int)(t3 & 1);
  int set = (int)(t3 >> 1);

  int l = sortbuf[(size_t)(set * 2 + b) * SEQ + j];
  int p = (set ? pc : pr)[b * SEQ + l];
  int fi = d & 31;
  float ang = (float)p * __expf(-(float)fi * 0.28782313662425572f);
  float ca = __cosf(ang), sa = __sinf(ang);

  size_t src = ((size_t)(b * SEQ + l)) * DMODEL + h * HDIM;
  float qv = q[src + d], qp = q[src + (d ^ 1)];
  float kv = k[src + d], kp = k[src + (d ^ 1)];
  float rq = (d & 1) ? qp : -qp;
  float rk = (d & 1) ? kp : -kp;

  size_t head = (((size_t)set * 2 + b) * NHEAD + h) * ((size_t)SEQ * HDIM);
  qs[head + (size_t)j * HDIM + d] = f2bf(qv * ca + rq * sa);
  ks[head + (size_t)j * HDIM + d] = f2bf(kv * ca + rk * sa);
  vt[head + (size_t)d * SEQ + j]  = f2bf(v[src + d]);
}

// ---------------------------------------------------------------------------
// Flash attention in sorted domain, causal, per-wave 16-row Q tile.
// K-blocks are double-buffered in LDS via async-to-LDS copies (ASYNCcnt).
// 4 waves per block: LDS = 4*(2*4KB K bufs) + 4*1KB P bufs = 36 KB.
// ---------------------------------------------------------------------------
__global__ __launch_bounds__(128) void attn_wmma(
    const unsigned short* __restrict__ qs, const unsigned short* __restrict__ ks,
    const unsigned short* __restrict__ vt, const int* __restrict__ sortbuf,
    float* __restrict__ outrc)
{
  __shared__ unsigned short pb[4][16 * 32];
#if USE_ASYNC
  __shared__ unsigned short kbuf[4][2][32 * HDIM];
#endif
  int w    = threadIdx.x >> 5;
  int lane = threadIdx.x & 31;
  int wid  = blockIdx.x * 4 + w;
  const int QT = SEQ / 16; // 128
  int qt  = wid % QT;
  int h   = (wid / QT) & (NHEAD - 1);
  int b   = (wid / (QT * NHEAD)) & 1;
  int set = wid / (QT * NHEAD * BATCH);
  int g8 = (lane >> 4) << 3;
  int r  = lane & 15;

  size_t base = (((size_t)set * 2 + b) * NHEAD + h) * ((size_t)SEQ * HDIM);
  const unsigned short* Q  = qs + base;
  const unsigned short* Kp = ks + base;
  const unsigned short* Vt = vt + base;
  const int* sp = sortbuf + (size_t)(set * 2 + b) * SEQ;
  int tq = qt * 16;

  // Q fragments (D = 64 -> two K-steps of 32)
  const unsigned short* qrow = Q + (size_t)(tq + r) * HDIM;
  v16bf qa0 = ld_frag(qrow + g8,      qrow + 16 + g8);
  v16bf qa1 = ld_frag(qrow + 32 + g8, qrow + 48 + g8);

  float rmax[8], rsum[8];
  v8f o0 = {}, o1 = {}, o2 = {}, o3 = {};
#pragma unroll
  for (int rr = 0; rr < 8; ++rr) { rmax[rr] = -3.0e38f; rsum[rr] = 0.f; }

  const float scale = 0.125f; // 1/sqrt(64)
  int nkb = (qt >> 1) + 1;
  int buf = 0;

#if USE_ASYNC
  stage_k_async(Kp, 0, &kbuf[w][0][0], lane);   // prologue: block 0
#endif

  for (int kb = 0; kb < nkb; ++kb) {
    int k0 = kb * 32;

#if USE_ASYNC
    // issue next block into the other buffer, then wait for current buffer
    if (kb + 1 < nkb) {
      stage_k_async(Kp, k0 + 32, &kbuf[w][buf ^ 1][0], lane);
      __builtin_amdgcn_s_wait_asynccnt(8);
    } else {
      __builtin_amdgcn_s_wait_asynccnt(0);
    }
    const unsigned short* kr0 = &kbuf[w][buf][(size_t)r * HDIM];
    const unsigned short* kr1 = &kbuf[w][buf][(size_t)(16 + r) * HDIM];
#else
    const unsigned short* kr0 = Kp + (size_t)(k0 + r) * HDIM;
    const unsigned short* kr1 = Kp + (size_t)(k0 + 16 + r) * HDIM;
    __builtin_prefetch(kr0 + 32 * HDIM, 0, 1);  // next key block
#endif

    v8f s0 = {}, s1 = {};
    {
      v16bf b0 = ld_frag(kr0 + g8, kr0 + 16 + g8);
      v16bf b1 = ld_frag(kr1 + g8, kr1 + 16 + g8);
      s0 = __builtin_amdgcn_wmma_f32_16x16x32_bf16(false, qa0, false, b0, (short)0, s0, false, false);
      s1 = __builtin_amdgcn_wmma_f32_16x16x32_bf16(false, qa0, false, b1, (short)0, s1, false, false);
      b0 = ld_frag(kr0 + 32 + g8, kr0 + 48 + g8);
      b1 = ld_frag(kr1 + 32 + g8, kr1 + 48 + g8);
      s0 = __builtin_amdgcn_wmma_f32_16x16x32_bf16(false, qa1, false, b0, (short)0, s0, false, false);
      s1 = __builtin_amdgcn_wmma_f32_16x16x32_bf16(false, qa1, false, b1, (short)0, s1, false, false);
    }

    // online softmax; C-layout: element rr of lane -> qrow g8+rr, keycol l%16
#pragma unroll
    for (int rr = 0; rr < 8; ++rr) {
      int qg = tq + g8 + rr;
      float a  = s0[rr] * scale; if (k0 + r      > qg) a  = -3.0e38f;
      float bb = s1[rr] * scale; if (k0 + 16 + r > qg) bb = -3.0e38f;
      float mx = fmaxf(a, bb);
      mx = fmaxf(mx, __shfl_xor(mx, 1, 16));
      mx = fmaxf(mx, __shfl_xor(mx, 2, 16));
      mx = fmaxf(mx, __shfl_xor(mx, 4, 16));
      mx = fmaxf(mx, __shfl_xor(mx, 8, 16));
      float nm = fmaxf(rmax[rr], mx);
      float cf = __expf(rmax[rr] - nm);
      float p0 = __expf(a - nm);
      float p1 = __expf(bb - nm);
      float ps = p0 + p1;
      ps += __shfl_xor(ps, 1, 16);
      ps += __shfl_xor(ps, 2, 16);
      ps += __shfl_xor(ps, 4, 16);
      ps += __shfl_xor(ps, 8, 16);
      rsum[rr] = rsum[rr] * cf + ps;
      rmax[rr] = nm;
      o0[rr] *= cf; o1[rr] *= cf; o2[rr] *= cf; o3[rr] *= cf;
      pb[w][(g8 + rr) * 32 + r]      = f2bf(p0);
      pb[w][(g8 + rr) * 32 + 16 + r] = f2bf(p1);
    }

    // re-fragment P (16x32) as WMMA A via per-wave LDS (same-wave DS ordering)
    v16bf pa = ld_frag(&pb[w][r * 32 + g8], &pb[w][r * 32 + 16 + g8]);

    // O(16x64) += P(16x32) @ V(32x64); vt rows make B-frags contiguous
#pragma unroll
    for (int c = 0; c < 4; ++c) {
      const unsigned short* vp = Vt + (size_t)(c * 16 + r) * SEQ + k0;
      v16bf vf = ld_frag(vp + g8, vp + 16 + g8);
      v8f& oc = (c == 0) ? o0 : (c == 1) ? o1 : (c == 2) ? o2 : o3;
      oc = __builtin_amdgcn_wmma_f32_16x16x32_bf16(false, pa, false, vf, (short)0, oc, false, false);
    }
    buf ^= 1;
  }

  // normalize + scatter back to original order
#pragma unroll
  for (int rr = 0; rr < 8; ++rr) {
    int j = tq + g8 + rr;
    int orig = sp[j];
    float inv = 1.f / rsum[rr];
    float* op = outrc + base + (size_t)orig * HDIM;
    op[r]      = o0[rr] * inv;
    op[16 + r] = o1[rr] * inv;
    op[32 + r] = o2[rr] * inv;
    op[48 + r] = o3[rr] * inv;
  }
}

// ---------------------------------------------------------------------------
// Router: w = softmax(silu(q@Wru+bru)@Wrd+brd); mix row/col heads.
// One wave per (b,h,l). Writes combined tensor directly as bf16.
// ---------------------------------------------------------------------------
__global__ __launch_bounds__(256) void router_combine(
    const float* __restrict__ q, const float* __restrict__ Wru,
    const float* __restrict__ bru, const float* __restrict__ Wrd,
    const float* __restrict__ brd, const float* __restrict__ outrc,
    unsigned short* __restrict__ combined)
{
  __shared__ float qsh[8][HDIM];
  int w = threadIdx.x >> 5, lane = threadIdx.x & 31;
  int wid = blockIdx.x * 8 + w;
  int l = wid % SEQ;
  int h = (wid / SEQ) & (NHEAD - 1);
  int b = wid / (SEQ * NHEAD);

  const float* qrow = q + ((size_t)(b * SEQ + l)) * DMODEL + h * HDIM;
  qsh[w][lane]      = qrow[lane];
  qsh[w][lane + 32] = qrow[lane + 32];

  float acc0 = 0.f, acc1 = 0.f;
#pragma unroll
  for (int jj = 0; jj < 8; ++jj) {
    int jn = lane * 8 + jj;
    float s = bru[jn];
    for (int d = 0; d < HDIM; ++d) s += qsh[w][d] * Wru[d * 256 + jn];
    float sil = s / (1.f + __expf(-s));
    acc0 += sil * Wrd[jn * 2 + 0];
    acc1 += sil * Wrd[jn * 2 + 1];
  }
#pragma unroll
  for (int off = 1; off < 32; off <<= 1) {
    acc0 += __shfl_xor(acc0, off, 32);
    acc1 += __shfl_xor(acc1, off, 32);
  }
  float l0 = acc0 + brd[0], l1 = acc1 + brd[1];
  float mx = fmaxf(l0, l1);
  float e0 = __expf(l0 - mx), e1 = __expf(l1 - mx);
  float inv = 1.f / (e0 + e1);
  float w0 = e0 * inv, w1 = e1 * inv;

  size_t ro = (((size_t)0 * 2 + b) * NHEAD + h) * ((size_t)SEQ * HDIM) + (size_t)l * HDIM;
  size_t co = (((size_t)1 * 2 + b) * NHEAD + h) * ((size_t)SEQ * HDIM) + (size_t)l * HDIM;
  size_t dst = ((size_t)(b * SEQ + l)) * DMODEL + h * HDIM;
  int d0 = lane * 2;
  combined[dst + d0]     = f2bf(w0 * outrc[ro + d0]     + w1 * outrc[co + d0]);
  combined[dst + d0 + 1] = f2bf(w0 * outrc[ro + d0 + 1] + w1 * outrc[co + d0 + 1]);
}

// ---------------------------------------------------------------------------
extern "C" void kernel_launch(void* const* d_in, const int* in_sizes, int n_in,
                              void* d_out, int out_size, void* d_ws, size_t ws_size,
                              hipStream_t stream) {
  const float* hs  = (const float*)d_in[0];
  const int*   pr  = (const int*)d_in[1];
  const int*   pc  = (const int*)d_in[2];
  const float* Wq  = (const float*)d_in[3];
  const float* bq  = (const float*)d_in[4];
  const float* Wk  = (const float*)d_in[5];
  const float* bk  = (const float*)d_in[6];
  const float* Wv  = (const float*)d_in[7];
  const float* bv  = (const float*)d_in[8];
  const float* Wo  = (const float*)d_in[9];
  const float* bo  = (const float*)d_in[10];
  const float* Wru = (const float*)d_in[11];
  const float* bru = (const float*)d_in[12];
  const float* Wrd = (const float*)d_in[13];
  const float* brd = (const float*)d_in[14];

  char* ws = (char*)d_ws;
  constexpr size_t SZ = (size_t)MROWS * DMODEL * sizeof(float); // 8 MB
  float* q     = (float*)(ws + 0 * SZ);
  float* k     = (float*)(ws + 1 * SZ);
  float* v     = (float*)(ws + 2 * SZ);
  float* outrc = (float*)(ws + 3 * SZ);                 // 2*SZ
  unsigned short* qsb  = (unsigned short*)(ws + 5 * SZ);
  unsigned short* ksb  = (unsigned short*)(ws + 6 * SZ);
  unsigned short* vtb  = (unsigned short*)(ws + 7 * SZ);
  unsigned short* hsb  = (unsigned short*)(ws + 8 * SZ);            // SZ/2
  unsigned short* comb = (unsigned short*)(ws + 8 * SZ + SZ / 2);   // SZ/2
  unsigned short* Wqt  = (unsigned short*)(ws + 9 * SZ);
  unsigned short* Wkt  = (unsigned short*)(ws + 9 * SZ +  512 * 1024);
  unsigned short* Wvt  = (unsigned short*)(ws + 9 * SZ + 1024 * 1024);
  unsigned short* Wot  = (unsigned short*)(ws + 9 * SZ + 1536 * 1024);
  int* sortbuf = (int*)(ws + 9 * SZ + 2048 * 1024);     // 32 KB

  const int NELT = MROWS * DMODEL;   // 2,097,152
  const int WELT = DMODEL * DMODEL;  //   262,144

  cvt_bf16  <<<NELT / 256, 256, 0, stream>>>(hs, hsb, NELT);
  cvt_bf16_t<<<WELT / 256, 256, 0, stream>>>(Wq, Wqt, DMODEL, DMODEL);
  cvt_bf16_t<<<WELT / 256, 256, 0, stream>>>(Wk, Wkt, DMODEL, DMODEL);
  cvt_bf16_t<<<WELT / 256, 256, 0, stream>>>(Wv, Wvt, DMODEL, DMODEL);
  cvt_bf16_t<<<WELT / 256, 256, 0, stream>>>(Wo, Wot, DMODEL, DMODEL);

  // (M/32)*(N/64) = 128*8 = 1024 waves -> 128 blocks
  gemm_bias_wmma<<<128, 256, 0, stream>>>(hsb, Wqt, bq, q, MROWS, DMODEL, DMODEL);
  gemm_bias_wmma<<<128, 256, 0, stream>>>(hsb, Wkt, bk, k, MROWS, DMODEL, DMODEL);
  gemm_bias_wmma<<<128, 256, 0, stream>>>(hsb, Wvt, bv, v, MROWS, DMODEL, DMODEL);

  sort_positions<<<4, 256, 0, stream>>>(pr, pc, sortbuf);

  rope_gather<<<16384, 256, 0, stream>>>(q, k, v, pr, pc, sortbuf, qsb, ksb, vtb);

  // 4096 waves, 4 waves/block -> 1024 blocks of 128
  attn_wmma<<<1024, 128, 0, stream>>>(qsb, ksb, vtb, sortbuf, outrc);

  router_combine<<<4096, 256, 0, stream>>>(q, Wru, bru, Wrd, brd, outrc, comb);

  gemm_bias_wmma<<<128, 256, 0, stream>>>(comb, Wot, bo, (float*)d_out, MROWS, DMODEL, DMODEL);
}
